// TransducerJoint_34144990003718
// MI455X (gfx1250) — matrix operand order repaired and checked by reading, not compiled, not run on hardware
//
#include <hip/hip_runtime.h>
#include <hip/hip_bf16.h>
#include <math.h>

// Problem constants (match reference)
constexpr int Bc = 4, Tc = 512, Uc = 128;
constexpr int Ec = 256, Pc = 256, Jc = 512, Vc = 1024;

typedef __attribute__((ext_vector_type(16))) _Float16 v16h;
typedef __attribute__((ext_vector_type(8)))  float    v8f;

// ---------------------------------------------------------------------------
// Kernel 1a: e = enc @ W_enc + b_enc  -> f32 workspace  [B*T, J]
// ---------------------------------------------------------------------------
__global__ __launch_bounds__(256) void proj_enc_kernel(
    const float* __restrict__ enc, const float* __restrict__ W,
    const float* __restrict__ bias, float* __restrict__ e_ws) {
  __shared__ float xrow[Ec];
  const int bt = blockIdx.x;              // 0 .. B*T-1
  const int tid = threadIdx.x;
  const float* x = enc + (size_t)bt * Ec;
  for (int i = tid; i < Ec; i += 256) xrow[i] = x[i];
  __syncthreads();
  for (int j = tid; j < Jc; j += 256) {
    float acc = bias[j];
#pragma unroll 4
    for (int k = 0; k < Ec; ++k) acc += xrow[k] * W[(size_t)k * Jc + j];
    e_ws[(size_t)bt * Jc + j] = acc;
  }
}

// ---------------------------------------------------------------------------
// Kernel 1b: d = dec @ W_dec (no bias) -> f32 workspace  [B*U, J]
// ---------------------------------------------------------------------------
__global__ __launch_bounds__(256) void proj_dec_kernel(
    const float* __restrict__ dec, const float* __restrict__ W,
    float* __restrict__ d_ws) {
  __shared__ float xrow[Pc];
  const int bu = blockIdx.x;              // 0 .. B*U-1
  const int tid = threadIdx.x;
  const float* x = dec + (size_t)bu * Pc;
  for (int i = tid; i < Pc; i += 256) xrow[i] = x[i];
  __syncthreads();
  for (int j = tid; j < Jc; j += 256) {
    float acc = 0.f;
#pragma unroll 4
    for (int k = 0; k < Pc; ++k) acc += xrow[k] * W[(size_t)k * Jc + j];
    d_ws[(size_t)bu * Jc + j] = acc;
  }
}

// ---------------------------------------------------------------------------
// Kernel 1c: pre-swizzle W_out [J,V] f32 into f16 WMMA B-fragment order.
// B tile (32x16 f16): lane L: n = L&15, hi = L>>4; half j (0..15): K=(j>>3)*16+hi*8+(j&7).
// Storage index: (((ntile*16 + kchunk)*32 + lane)*16 + j) -> 32 contiguous
// bytes per lane per fragment => 2x global_load_b128, coalesced.
// ---------------------------------------------------------------------------
__global__ __launch_bounds__(256) void swizzle_wout_kernel(
    const float* __restrict__ W_out, _Float16* __restrict__ w16s) {
  const int idx = blockIdx.x * 256 + threadIdx.x;   // 0 .. 64*16*32*16-1
  const int j    = idx & 15;
  const int lane = (idx >> 4) & 31;
  const int kc   = (idx >> 9) & 15;
  const int nt   = idx >> 13;                       // 0..63
  const int hi = lane >> 4;
  const int n  = nt * 16 + (lane & 15);
  const int k  = kc * 32 + (j >> 3) * 16 + hi * 8 + (j & 7);
  w16s[idx] = (_Float16)W_out[(size_t)k * Vc + n];
}

// ---------------------------------------------------------------------------
// Kernel 2 (main): fused tanh(e+d) -> WMMA GEMM vs W_out -> +bias ->
// log_softmax -> streaming stores.
// Block = (b, t, 32 consecutive u). 8 waves; wave w owns cols [128w,128w+128).
// Two M-tiles of 16 rows each => every B fragment feeds 2 WMMAs (halves L2
// traffic for W_out vs a 16-row block).
// ---------------------------------------------------------------------------
__global__ __launch_bounds__(256) void joint_main_kernel(
    const float* __restrict__ e_ws, const float* __restrict__ d_ws,
    const _Float16* __restrict__ w16s, const float* __restrict__ b_out,
    float* __restrict__ out) {
  __shared__ _Float16 hA[32 * Jc];        // 32 KB: A tiles, f16(tanh(e+d))
  __shared__ float red[2][8][32];         // cross-wave partials (max / sum)

  const int tid = threadIdx.x;
  const int blk = blockIdx.x;             // B*T*(U/32) = 8192
  const int ut = blk & 3;
  const int t  = (blk >> 2) & (Tc - 1);
  const int b  = blk >> 11;               // / (T*4)
  const int u0 = ut * 32;

  const float* erow  = e_ws + ((size_t)(b * Tc + t)) * Jc;
  const float* drows = d_ws + ((size_t)(b * Uc + u0)) * Jc;

  // ---- Build A tiles in LDS: hA[m*J + k] = f16(tanh(e[k] + d[m][k])) ----
  for (int idx = tid; idx < 32 * Jc; idx += 256) {
    const int m = idx >> 9;               // J = 512
    const int k = idx & (Jc - 1);
    const float x = erow[k] + drows[(size_t)m * Jc + k];
    float th;
    // CDNA5 transcendental; v_nop covers the 1-op trans result hazard.
    asm volatile("v_tanh_f32 %0, %1\n\tv_nop" : "=v"(th) : "v"(x));
    hA[idx] = (_Float16)th;
  }
  __syncthreads();

  const int w    = tid >> 5;              // wave id 0..7
  const int lane = tid & 31;
  const int hi   = lane >> 4;
  const int ln   = lane & 15;
  const int wbase = w * 128;              // first vocab column of this wave

  v8f acc0[8], acc1[8];
#pragma unroll
  for (int nt = 0; nt < 8; ++nt)
#pragma unroll
    for (int i = 0; i < 8; ++i) { acc0[nt][i] = 0.f; acc1[nt][i] = 0.f; }

  // Hoisted per-lane base pointers: all loop offsets are compile-time
  // constants -> immediate offsets on ds_load_b128 / global_load_b128.
  const _Float16* aB0 = &hA[(size_t)ln * Jc + hi * 8];        // rows 0..15
  const _Float16* aB1 = aB0 + 16 * Jc;                        // rows 16..31
  const _Float16* bB  = w16s + ((size_t)(w * 8) * 16 * 32 + lane) * 16;

  // ---- K loop: 16 chunks of 32 ----
#pragma unroll 2
  for (int kc = 0; kc < 16; ++kc) {
    // A fragment (16-bit A 16x32 layout): m = ln; elems 0..7 -> K=hi*8+p,
    // elems 8..15 -> K=16+hi*8+p (within this 32-wide chunk).
    v16h a0, a1;
#pragma unroll
    for (int p = 0; p < 8; ++p) {
      a0[p]     = aB0[kc * 32 + p];
      a0[8 + p] = aB0[kc * 32 + 16 + p];
      a1[p]     = aB1[kc * 32 + p];
      a1[8 + p] = aB1[kc * 32 + 16 + p];
    }
#pragma unroll
    for (int nt = 0; nt < 8; ++nt) {
      const v16h bfrag = *(const v16h*)(bB + (nt * 16 + kc) * 512);
      acc0[nt] = __builtin_amdgcn_wmma_f32_16x16x32_f16(
          false, a0, false, bfrag, (short)0, acc0[nt], false, false);
      acc1[nt] = __builtin_amdgcn_wmma_f32_16x16x32_f16(
          false, a1, false, bfrag, (short)0, acc1[nt], false, false);
    }
  }

  // ---- bias ----
  float bo[8];
#pragma unroll
  for (int nt = 0; nt < 8; ++nt) bo[nt] = b_out[wbase + nt * 16 + ln];
#pragma unroll
  for (int nt = 0; nt < 8; ++nt)
#pragma unroll
    for (int i = 0; i < 8; ++i) { acc0[nt][i] += bo[nt]; acc1[nt][i] += bo[nt]; }

  // C/D layout: lane col = wbase + nt*16 + ln; VGPR i -> row mt*16 + i + 8*hi.
  // ---- per-row max over this wave's 128 cols (reduce 16-lane column group)
  float rmax0[8], rmax1[8];
#pragma unroll
  for (int i = 0; i < 8; ++i) {
    float m0 = acc0[0][i], m1 = acc1[0][i];
#pragma unroll
    for (int nt = 1; nt < 8; ++nt) {
      m0 = fmaxf(m0, acc0[nt][i]);
      m1 = fmaxf(m1, acc1[nt][i]);
    }
#pragma unroll
    for (int off = 1; off < 16; off <<= 1) {   // masks 1,2,4,8 stay in half
      m0 = fmaxf(m0, __shfl_xor(m0, off, 32));
      m1 = fmaxf(m1, __shfl_xor(m1, off, 32));
    }
    rmax0[i] = m0; rmax1[i] = m1;
  }
  if (ln == 0) {
#pragma unroll
    for (int i = 0; i < 8; ++i) {
      red[0][w][i + 8 * hi]      = rmax0[i];
      red[0][w][16 + i + 8 * hi] = rmax1[i];
    }
  }
  __syncthreads();
  float rowmax0[8], rowmax1[8];
#pragma unroll
  for (int i = 0; i < 8; ++i) {
    const int r = i + 8 * hi;
    float m0 = red[0][0][r], m1 = red[0][0][16 + r];
#pragma unroll
    for (int ww = 1; ww < 8; ++ww) {
      m0 = fmaxf(m0, red[0][ww][r]);
      m1 = fmaxf(m1, red[0][ww][16 + r]);
    }
    rowmax0[i] = m0; rowmax1[i] = m1;
  }

  // ---- per-row sum(exp(x - max)) ----
  float rsum0[8], rsum1[8];
#pragma unroll
  for (int i = 0; i < 8; ++i) {
    float s0 = 0.f, s1 = 0.f;
#pragma unroll
    for (int nt = 0; nt < 8; ++nt) {
      s0 += __expf(acc0[nt][i] - rowmax0[i]);
      s1 += __expf(acc1[nt][i] - rowmax1[i]);
    }
#pragma unroll
    for (int off = 1; off < 16; off <<= 1) {
      s0 += __shfl_xor(s0, off, 32);
      s1 += __shfl_xor(s1, off, 32);
    }
    rsum0[i] = s0; rsum1[i] = s1;
  }
  if (ln == 0) {
#pragma unroll
    for (int i = 0; i < 8; ++i) {
      red[1][w][i + 8 * hi]      = rsum0[i];
      red[1][w][16 + i + 8 * hi] = rsum1[i];
    }
  }
  __syncthreads();
  float lse0[8], lse1[8];
#pragma unroll
  for (int i = 0; i < 8; ++i) {
    const int r = i + 8 * hi;
    float s0 = 0.f, s1 = 0.f;
#pragma unroll
    for (int ww = 0; ww < 8; ++ww) {
      s0 += red[1][ww][r];
      s1 += red[1][ww][16 + r];
    }
    lse0[i] = rowmax0[i] + __logf(s0);
    lse1[i] = rowmax1[i] + __logf(s1);
  }

  // ---- streaming stores: out[b,t,u0+row, :] (32x1024 contiguous tile) ----
  const size_t base = (((size_t)(b * Tc + t) * Uc) + u0) * Vc;
#pragma unroll
  for (int i = 0; i < 8; ++i) {
    const int r = i + 8 * hi;
    float* orow0 = out + base + (size_t)r * Vc;
    float* orow1 = orow0 + (size_t)16 * Vc;
#pragma unroll
    for (int nt = 0; nt < 8; ++nt) {
      __builtin_nontemporal_store(acc0[nt][i] - lse0[i],
                                  &orow0[wbase + nt * 16 + ln]);
      __builtin_nontemporal_store(acc1[nt][i] - lse1[i],
                                  &orow1[wbase + nt * 16 + ln]);
    }
  }
}

// ---------------------------------------------------------------------------
// Host-side launcher
// ---------------------------------------------------------------------------
extern "C" void kernel_launch(void* const* d_in, const int* in_sizes, int n_in,
                              void* d_out, int out_size, void* d_ws, size_t ws_size,
                              hipStream_t stream) {
  (void)in_sizes; (void)n_in; (void)out_size; (void)ws_size;
  const float* enc   = (const float*)d_in[0];   // [B,T,E]
  const float* dec   = (const float*)d_in[1];   // [B,U,P]
  const float* W_enc = (const float*)d_in[2];   // [E,J]
  const float* b_enc = (const float*)d_in[3];   // [J]
  const float* W_dec = (const float*)d_in[4];   // [P,J]
  const float* W_out = (const float*)d_in[5];   // [J,V]
  const float* b_out = (const float*)d_in[6];   // [V]
  float* out = (float*)d_out;                   // [B,T,U,V]

  // Workspace layout (needs 6 MB):
  //   e_ws  : B*T*J f32  (4 MB)
  //   d_wsp : B*U*J f32  (1 MB)
  //   w16s  : J*V   f16  (1 MB, WMMA-B swizzled)
  float* e_ws  = (float*)d_ws;
  float* d_wsp = e_ws + (size_t)Bc * Tc * Jc;
  _Float16* w16s = (_Float16*)(d_wsp + (size_t)Bc * Uc * Jc);

  proj_enc_kernel<<<Bc * Tc, 256, 0, stream>>>(enc, W_enc, b_enc, e_ws);
  proj_dec_kernel<<<Bc * Uc, 256, 0, stream>>>(dec, W_dec, d_wsp);
  swizzle_wout_kernel<<<(64 * 16 * 32 * 16) / 256, 256, 0, stream>>>(W_out, w16s);
  joint_main_kernel<<<Bc * Tc * (Uc / 32), 256, 0, stream>>>(
      e_ws, d_wsp, w16s, b_out, out);
}
